// EncoderLayer_28467043238446
// MI455X (gfx1250) — matrix operand order repaired and checked by reading, not compiled, not run on hardware
//
#include <hip/hip_runtime.h>
#include <hip/hip_bf16.h>

// ---------------- CDNA5 WMMA types ----------------
typedef __attribute__((ext_vector_type(16))) __bf16 v16bf;
typedef __attribute__((ext_vector_type(8)))  __bf16 v8bf;
typedef __attribute__((ext_vector_type(8)))  float  v8f;

#define BM 128
#define BN 64
#define BK 32
#define LDA 40   // BK + 8 pad (bf16 elems) -> 80B row pitch, conflict-free & 16B aligned

static __device__ __forceinline__ v16bf make_v16(v8bf lo, v8bf hi) {
  return __builtin_shufflevector(lo, hi, 0,1,2,3,4,5,6,7,8,9,10,11,12,13,14,15);
}

// gfx1250 async LDS copy: 16 bytes per lane, tracked by ASYNCcnt (in-order).
static __device__ __forceinline__ void async_ld16(unsigned lds_off, const void* gaddr) {
  asm volatile("global_load_async_to_lds_b128 %0, %1, off"
               :: "v"(lds_off), "v"(gaddr) : "memory");
}
static __device__ __forceinline__ void wait_async0() {
  asm volatile("s_wait_asynccnt 0x0" ::: "memory");
}
static __device__ __forceinline__ void wait_async3() {   // current tile done, next in flight
  asm volatile("s_wait_asynccnt 0x3" ::: "memory");
}
static __device__ __forceinline__ void wait_async4() {
  asm volatile("s_wait_asynccnt 0x4" ::: "memory");
}

// ---------------------------------------------------------------------------
// fp32 -> bf16 bulk convert (4 elems/thread)
// ---------------------------------------------------------------------------
__global__ __launch_bounds__(256) void cvt_bf16(
    const float* __restrict__ in, __bf16* __restrict__ out)
{
  const int i = (blockIdx.x * 256 + threadIdx.x) * 4;
  float4 v = *(const float4*)(in + i);
  union { __bf16 h[4]; uint2 u; } t;
  t.h[0] = (__bf16)v.x; t.h[1] = (__bf16)v.y;
  t.h[2] = (__bf16)v.z; t.h[3] = (__bf16)v.w;
  *(uint2*)(out + i) = t.u;
}

// ---------------------------------------------------------------------------
// Weight transpose + convert: W[K,N] fp32 -> Wt[N,K] bf16 (64x64 LDS tiles)
// ---------------------------------------------------------------------------
__global__ __launch_bounds__(256) void wtrans(
    const float* __restrict__ W, __bf16* __restrict__ Wt, int K, int N)
{
  __shared__ __bf16 t[64 * 65];
  const int tid = threadIdx.x;
  const int n0 = blockIdx.x * 64, k0 = blockIdx.y * 64;
  for (int l = tid; l < 4096; l += 256) {
    const int kk = l >> 6, nn = l & 63;
    t[nn * 65 + kk] = (__bf16)W[(size_t)(k0 + kk) * N + n0 + nn];
  }
  __syncthreads();
  for (int l = tid; l < 4096; l += 256) {
    const int nn = l >> 6, kk = l & 63;
    Wt[(size_t)(n0 + nn) * K + k0 + kk] = t[nn * 65 + kk];
  }
}

// ---------------------------------------------------------------------------
// GEMM: C = act(A[M,K](bf16) @ Wt[N,K](bf16)^T + bias), f32 WMMA accum.
// Double-buffered GLOBAL_LOAD_ASYNC_TO_LDS_B128 staging: copies for tile k+1
// overlap WMMA compute on tile k (ASYNCcnt completes in order, so
// s_wait_asynccnt<=3 releases the current tile while the next is in flight).
// Block 256 thr (8 waves), tile 128x64, each wave -> 16 rows x 64 cols.
// ---------------------------------------------------------------------------
__global__ __launch_bounds__(256) void gemm_bf16(
    const __bf16* __restrict__ A, const __bf16* __restrict__ Wt,
    const float* __restrict__ bias, float* __restrict__ Cf,
    __bf16* __restrict__ Cbf, int Mdim, int Ndim, int Kdim, int act)
{
  __shared__ __bf16 Ash[2][BM * LDA];
  __shared__ __bf16 Wsh[2][BN * LDA];

  const int tid  = threadIdx.x;
  const int wave = tid >> 5;
  const int lane = tid & 31;
  const int n0 = blockIdx.x * BN;
  const int m0 = blockIdx.y * BM;

  unsigned aB[2] = { (unsigned)(size_t)&Ash[0][0], (unsigned)(size_t)&Ash[1][0] };
  unsigned wB[2] = { (unsigned)(size_t)&Wsh[0][0], (unsigned)(size_t)&Wsh[1][0] };

  v8f acc[4];
  for (int j = 0; j < 4; ++j)
    for (int r = 0; r < 8; ++r) acc[j][r] = 0.0f;

  // staging decomposition: 16B chunks, A: 128 rows x 4, W: 64 rows x 4
  const int ar0 = tid >> 2, ar1 = 64 + (tid >> 2), aci = tid & 3;

  const int am  = wave * 16 + (lane & 15);
  const int akb = (lane >> 4) * 8;
  const int bkb = (lane < 16) ? 0 : 8;
  const int bnl = lane & 15;

  // prologue: stage tile 0 into buffer 0
  async_ld16(aB[0] + ar0 * 80 + aci * 16, A + (size_t)(m0 + ar0) * Kdim + aci * 8);
  async_ld16(aB[0] + ar1 * 80 + aci * 16, A + (size_t)(m0 + ar1) * Kdim + aci * 8);
  async_ld16(wB[0] + ar0 * 80 + aci * 16, Wt + (size_t)(n0 + ar0) * Kdim + aci * 8);

  int cur = 0;
  for (int kk = 0; kk < Kdim; kk += BK) {
    const int kn = kk + BK;
    if (kn < Kdim) {  // issue next tile into the other buffer, then release current
      const int nb = cur ^ 1;
      async_ld16(aB[nb] + ar0 * 80 + aci * 16, A + (size_t)(m0 + ar0) * Kdim + kn + aci * 8);
      async_ld16(aB[nb] + ar1 * 80 + aci * 16, A + (size_t)(m0 + ar1) * Kdim + kn + aci * 8);
      async_ld16(wB[nb] + ar0 * 80 + aci * 16, Wt + (size_t)(n0 + ar0) * Kdim + kn + aci * 8);
      if (kn + BK < Kdim) {  // gfx1250 global_prefetch_b8 two tiles ahead
        __builtin_prefetch(A + (size_t)(m0 + ar0) * Kdim + kn + BK + aci * 8, 0, 0);
        __builtin_prefetch(Wt + (size_t)(n0 + ar0) * Kdim + kn + BK + aci * 8, 0, 0);
      }
      wait_async3();
    } else {
      wait_async0();
    }
    __syncthreads();

    // A fragment (16x32 bf16, wave32 layout)
    v8bf alo = *(const v8bf*)&Ash[cur][am * LDA + akb];
    v8bf ahi = *(const v8bf*)&Ash[cur][am * LDA + akb + 16];
    v16bf afrag = make_v16(alo, ahi);

    // preload all 4 B fragments, then back-to-back WMMAs
    v16bf bfr[4];
    for (int j = 0; j < 4; ++j) {
      const int bn = j * 16 + bnl;
      v8bf blo = *(const v8bf*)&Wsh[cur][bn * LDA + bkb];
      v8bf bhi = *(const v8bf*)&Wsh[cur][bn * LDA + bkb + 16];
      bfr[j] = make_v16(blo, bhi);
    }
    for (int j = 0; j < 4; ++j)
      acc[j] = __builtin_amdgcn_wmma_f32_16x16x32_bf16(
          false, afrag, false, bfr[j], (short)0, acc[j], false, false);
    __syncthreads();   // all waves done reading before buffer is overwritten
    cur ^= 1;
  }

  // epilogue: bias + activation + store (C/D layout)
  const int mbase = m0 + wave * 16 + ((lane >> 4) << 3);
  for (int j = 0; j < 4; ++j) {
    const int ncol = n0 + j * 16 + bnl;
    const float bvl = bias[ncol];
    for (int r = 0; r < 8; ++r) {
      float v = acc[j][r] + bvl;
      if (act == 1) v = 0.5f * v * (1.0f + erff(v * 0.70710678118f));
      const size_t off = (size_t)(mbase + r) * Ndim + ncol;
      if (Cf)  Cf[off]  = v;
      if (Cbf) Cbf[off] = (__bf16)v;
    }
  }
}

// ---------------------------------------------------------------------------
// Head-mean circular autocorrelation via WMMA Gram tiles (bf16 Q/K inputs).
// mean_corr[bh,tau] += (1/64) * sum_{i-j==tau (mod 512)} Q[i,:] . K[j,:]
// Both d-chunks issued up front (double-buffered async).
// ---------------------------------------------------------------------------
__global__ __launch_bounds__(256) void corr_wmma(
    const __bf16* __restrict__ Q, const __bf16* __restrict__ Km,
    float* __restrict__ corr)
{
  __shared__ __bf16 Qsh[2][128 * LDA];
  __shared__ __bf16 Ksh[2][128 * LDA];
  __shared__ float  part[512];

  const int tid  = threadIdx.x;
  const int wave = tid >> 5;
  const int lane = tid & 31;
  const int j0 = blockIdx.x * 128;
  const int i0 = blockIdx.y * 128;
  const int bh = blockIdx.z;
  const int b  = bh >> 3, h = bh & 7;

  const __bf16* Qb = Q  + ((size_t)b * 512) * 512 + h * 64;
  const __bf16* Kb = Km + ((size_t)b * 512) * 512 + h * 64;

  unsigned qB[2] = { (unsigned)(size_t)&Qsh[0][0], (unsigned)(size_t)&Qsh[1][0] };
  unsigned kB[2] = { (unsigned)(size_t)&Ksh[0][0], (unsigned)(size_t)&Ksh[1][0] };

  part[tid] = 0.0f; part[tid + 256] = 0.0f;

  v8f acc[8];
  for (int j = 0; j < 8; ++j)
    for (int r = 0; r < 8; ++r) acc[j][r] = 0.0f;

  const int sr0 = tid >> 2, sr1 = 64 + (tid >> 2), sci = tid & 3;
  const int am  = wave * 16 + (lane & 15);
  const int akb = (lane >> 4) * 8;
  const int bkb = (lane < 16) ? 0 : 8;
  const int bnl = lane & 15;

  // issue both 32-wide d-chunks up front
  for (int half = 0; half < 2; ++half) {
    const int dk = half * 32;
    async_ld16(qB[half] + sr0 * 80 + sci * 16, Qb + (size_t)(i0 + sr0) * 512 + dk + sci * 8);
    async_ld16(qB[half] + sr1 * 80 + sci * 16, Qb + (size_t)(i0 + sr1) * 512 + dk + sci * 8);
    async_ld16(kB[half] + sr0 * 80 + sci * 16, Kb + (size_t)(j0 + sr0) * 512 + dk + sci * 8);
    async_ld16(kB[half] + sr1 * 80 + sci * 16, Kb + (size_t)(j0 + sr1) * 512 + dk + sci * 8);
  }

  for (int half = 0; half < 2; ++half) {
    if (half == 0) wait_async4(); else wait_async0();
    __syncthreads();

    v8bf alo = *(const v8bf*)&Qsh[half][am * LDA + akb];
    v8bf ahi = *(const v8bf*)&Qsh[half][am * LDA + akb + 16];
    v16bf afrag = make_v16(alo, ahi);

    v16bf bfr[8];
    for (int jt = 0; jt < 8; ++jt) {
      const int bn = jt * 16 + bnl;
      v8bf blo = *(const v8bf*)&Ksh[half][bn * LDA + bkb];
      v8bf bhi = *(const v8bf*)&Ksh[half][bn * LDA + bkb + 16];
      bfr[jt] = make_v16(blo, bhi);
    }
    for (int jt = 0; jt < 8; ++jt)
      acc[jt] = __builtin_amdgcn_wmma_f32_16x16x32_bf16(
          false, afrag, false, bfr[jt], (short)0, acc[jt], false, false);
  }
  __syncthreads();

  // diagonal accumulation: tau = (i - j) mod 512
  const int ibase = i0 + wave * 16 + ((lane >> 4) << 3);
  for (int jt = 0; jt < 8; ++jt) {
    const int j = j0 + jt * 16 + bnl;
    for (int r = 0; r < 8; ++r) {
      const int tau = ((ibase + r) - j) & 511;
      atomicAdd(&part[tau], acc[jt][r]);
    }
  }
  __syncthreads();

  float s0 = part[tid];
  if (s0 != 0.0f) atomicAdd(&corr[(size_t)bh * 512 + tid], s0 * (1.0f / 64.0f));
  float s1 = part[tid + 256];
  if (s1 != 0.0f) atomicAdd(&corr[(size_t)bh * 512 + tid + 256], s1 * (1.0f / 64.0f));
}

// ---------------------------------------------------------------------------
// Top-6 (value, lag) per (b,h) over 512 lags. One block per bh.
// ---------------------------------------------------------------------------
__global__ __launch_bounds__(256) void topk6(
    const float* __restrict__ corr, float* __restrict__ vals, int* __restrict__ lags)
{
  __shared__ float sv[512];
  __shared__ float rv[256];
  __shared__ int   ri[256];
  const int bh = blockIdx.x, tid = threadIdx.x;
  sv[tid]       = corr[(size_t)bh * 512 + tid];
  sv[tid + 256] = corr[(size_t)bh * 512 + tid + 256];
  __syncthreads();
  for (int it = 0; it < 6; ++it) {
    float v0 = sv[tid], v1 = sv[tid + 256];
    float mv; int mi;
    if (v0 >= v1) { mv = v0; mi = tid; } else { mv = v1; mi = tid + 256; }
    rv[tid] = mv; ri[tid] = mi;
    __syncthreads();
    for (int off = 128; off > 0; off >>= 1) {
      if (tid < off && rv[tid + off] > rv[tid]) { rv[tid] = rv[tid + off]; ri[tid] = ri[tid + off]; }
      __syncthreads();
    }
    if (tid == 0) {
      vals[bh * 6 + it] = rv[0];
      lags[bh * 6 + it] = ri[0];
      sv[ri[0]] = -3.4e38f;
    }
    __syncthreads();
  }
}

// ---------------------------------------------------------------------------
// Global mean lag (floor) per k-slot + per-(b,h) softmax weights. One block.
// ---------------------------------------------------------------------------
__global__ __launch_bounds__(256) void finalize_wl(
    const float* __restrict__ vals, const int* __restrict__ lags,
    float* __restrict__ wts, int* __restrict__ lagint)
{
  __shared__ float red[256];
  const int tid = threadIdx.x;
  for (int i = 0; i < 6; ++i) {
    red[tid] = (float)lags[tid * 6 + i];
    __syncthreads();
    for (int off = 128; off > 0; off >>= 1) {
      if (tid < off) red[tid] += red[tid + off];
      __syncthreads();
    }
    if (tid == 0) lagint[i] = (int)floorf(red[0] * (1.0f / 256.0f));
    __syncthreads();
  }
  float v[6], mx = -3.4e38f;
  for (int i = 0; i < 6; ++i) { v[i] = vals[tid * 6 + i]; mx = fmaxf(mx, v[i]); }
  float se = 0.0f;
  for (int i = 0; i < 6; ++i) { v[i] = __expf(v[i] - mx); se += v[i]; }
  const float inv = 1.0f / se;
  for (int i = 0; i < 6; ++i) wts[tid * 6 + i] = v[i] * inv;
}

// ---------------------------------------------------------------------------
// out[b,t,c] = sum_i w[b,h(c),i] * V[b,(t+lag_i) mod 512, c]   (bf16 out)
// ---------------------------------------------------------------------------
__global__ __launch_bounds__(256) void combine(
    const float* __restrict__ V, const float* __restrict__ wts,
    const int* __restrict__ lagint, __bf16* __restrict__ out)
{
  const int idx = blockIdx.x * 256 + threadIdx.x;
  const int c = idx & 511;
  const int t = (idx >> 9) & 511;
  const int b = idx >> 18;
  const int bh = (b << 3) + (c >> 6);
  float s = 0.0f;
  for (int i = 0; i < 6; ++i) {
    const int tt = (t + lagint[i]) & 511;
    s += wts[bh * 6 + i] * V[(((size_t)b << 9) + tt) * 512 + c];
  }
  out[idx] = (__bf16)s;
}

// ---------------------------------------------------------------------------
// LayerNorm(res + add) * g + be, one wave32 per 512-wide row.
// ---------------------------------------------------------------------------
__global__ __launch_bounds__(256) void ln_resid(
    const float* __restrict__ res, const float* __restrict__ add,
    const float* __restrict__ g, const float* __restrict__ be,
    float* __restrict__ out)
{
  const int tid = threadIdx.x, wave = tid >> 5, lane = tid & 31;
  const size_t row = (size_t)blockIdx.x * 8 + wave;
  const float* r0 = res + row * 512;
  const float* a0 = add + row * 512;
  float e[16], s = 0.0f, ss = 0.0f;
  for (int j = 0; j < 16; ++j) {
    e[j] = r0[lane + j * 32] + a0[lane + j * 32];
    s += e[j]; ss += e[j] * e[j];
  }
  for (int off = 16; off > 0; off >>= 1) {
    s  += __shfl_xor(s,  off, 32);
    ss += __shfl_xor(ss, off, 32);
  }
  const float mu  = s * (1.0f / 512.0f);
  const float var = ss * (1.0f / 512.0f) - mu * mu;
  const float rstd = rsqrtf(var + 1e-5f);
  for (int j = 0; j < 16; ++j) {
    const int c = lane + j * 32;
    out[row * 512 + c] = (e[j] - mu) * rstd * g[c] + be[c];
  }
}

// ---------------------------------------------------------------------------
// series_decomp seasonal: out = y - movavg25(edge-padded y) along T.
// Optional bf16 copy for feeding the next GEMM.
// ---------------------------------------------------------------------------
__global__ __launch_bounds__(256) void decomp(
    const float* __restrict__ y, float* __restrict__ outp,
    __bf16* __restrict__ outbf)
{
  __shared__ float tile[88 * 64];
  const int tid = threadIdx.x;
  const int c0 = blockIdx.x * 64, t0 = blockIdx.y * 64, b = blockIdx.z;
  for (int l = tid; l < 88 * 64; l += 256) {
    const int rr = l >> 6, cc = l & 63;
    int tg = t0 + rr - 12;
    tg = tg < 0 ? 0 : (tg > 511 ? 511 : tg);
    tile[l] = y[(((size_t)b << 9) + tg) * 512 + c0 + cc];
  }
  __syncthreads();
  for (int q = 0; q < 16; ++q) {
    const int o = tid + q * 256;
    const int tl = o >> 6, cc = o & 63;
    float s = 0.0f;
    for (int w = 0; w < 25; ++w) s += tile[(tl + w) * 64 + cc];
    const float v = tile[(tl + 12) * 64 + cc] - s * (1.0f / 25.0f);
    const size_t off = (((size_t)b << 9) + (t0 + tl)) * 512 + c0 + cc;
    outp[off] = v;
    if (outbf) outbf[off] = (__bf16)v;
  }
}

// ---------------------------------------------------------------------------
extern "C" void kernel_launch(void* const* d_in, const int* in_sizes, int n_in,
                              void* d_out, int out_size, void* d_ws, size_t ws_size,
                              hipStream_t stream)
{
  (void)in_sizes; (void)n_in; (void)out_size; (void)ws_size;
  const float* x   = (const float*)d_in[0];
  const float* Wq  = (const float*)d_in[1];  const float* bq  = (const float*)d_in[2];
  const float* Wk  = (const float*)d_in[3];  const float* bk  = (const float*)d_in[4];
  const float* Wv  = (const float*)d_in[5];  const float* bv  = (const float*)d_in[6];
  const float* Wo  = (const float*)d_in[7];  const float* bo  = (const float*)d_in[8];
  const float* W1  = (const float*)d_in[9];  const float* b1  = (const float*)d_in[10];
  const float* W2  = (const float*)d_in[11]; const float* b2  = (const float*)d_in[12];
  const float* g1  = (const float*)d_in[13]; const float* be1 = (const float*)d_in[14];
  const float* g2  = (const float*)d_in[15]; const float* be2 = (const float*)d_in[16];
  float* out = (float*)d_out;

  const size_t MD = 16384ull * 512;          // 8,388,608
  float* WS   = (float*)d_ws;
  float*  bufA = WS;                         // V fp32, later f
  float*  bufB = WS + MD;                    // a, later h fp32
  float*  bufC = WS + 2 * MD;                // y1, y2
  __bf16* bigbf = (__bf16*)(WS + 3 * MD);    // 16384x2048 bf16 (gelu out)
  __bf16* xbf   = (__bf16*)(WS + 5 * MD);    // MD bf16
  __bf16* Qbf   = (__bf16*)(WS + 5 * MD + MD / 2);
  __bf16* Kbf   = (__bf16*)(WS + 6 * MD);
  __bf16* abf   = (__bf16*)(WS + 6 * MD + MD / 2);   // AC bf16, later h bf16
  float*  wb    = WS + 7 * MD;
  __bf16* Wqt = (__bf16*)(wb);
  __bf16* Wkt = (__bf16*)(wb + 131072);
  __bf16* Wvt = (__bf16*)(wb + 2 * 131072);
  __bf16* Wot = (__bf16*)(wb + 3 * 131072);
  __bf16* W1t = (__bf16*)(wb + 4 * 131072);            // 2048x512 bf16
  __bf16* W2t = (__bf16*)(wb + 4 * 131072 + 524288);   // 512x2048 bf16
  float* smal = wb + 4 * 131072 + 2 * 524288;
  float* corr = smal;                        // 256*512
  float* vals = corr + 131072;               // 256*6
  int*   lags = (int*)(vals + 1536);         // 256*6
  float* wts  = (float*)(lags + 1536);       // 256*6
  int*   lagi = (int*)(wts + 1536);          // 6

  const dim3 blk(256);
  const dim3 g512(512 / BN, 16384 / BM);     // (8,128)

  // one-time bf16 conversion of x and transposed weights
  cvt_bf16<<<dim3(MD / 1024), blk, 0, stream>>>(x, xbf);
  wtrans<<<dim3(8, 8),  blk, 0, stream>>>(Wq, Wqt, 512, 512);
  wtrans<<<dim3(8, 8),  blk, 0, stream>>>(Wk, Wkt, 512, 512);
  wtrans<<<dim3(8, 8),  blk, 0, stream>>>(Wv, Wvt, 512, 512);
  wtrans<<<dim3(8, 8),  blk, 0, stream>>>(Wo, Wot, 512, 512);
  wtrans<<<dim3(32, 8), blk, 0, stream>>>(W1, W1t, 512, 2048);
  wtrans<<<dim3(8, 32), blk, 0, stream>>>(W2, W2t, 2048, 512);

  // Q, K, V projections (Q,K -> bf16 for correlation; V -> fp32 for combine)
  gemm_bf16<<<g512, blk, 0, stream>>>(xbf, Wqt, bq, nullptr, Qbf, 16384, 512, 512, 0);
  gemm_bf16<<<g512, blk, 0, stream>>>(xbf, Wkt, bk, nullptr, Kbf, 16384, 512, 512, 0);
  gemm_bf16<<<g512, blk, 0, stream>>>(xbf, Wvt, bv, bufA, nullptr, 16384, 512, 512, 0);

  // head-mean circular autocorrelation
  hipMemsetAsync(corr, 0, 131072 * sizeof(float), stream);
  corr_wmma<<<dim3(4, 4, 256), blk, 0, stream>>>(Qbf, Kbf, corr);
  topk6<<<dim3(256), blk, 0, stream>>>(corr, vals, lags);
  finalize_wl<<<dim3(1), blk, 0, stream>>>(vals, lags, wts, lagi);
  combine<<<dim3(32768), blk, 0, stream>>>(bufA, wts, lagi, abf);   // AC -> abf (bf16)

  // output projection + LN1 + decomp1
  gemm_bf16<<<g512, blk, 0, stream>>>(abf, Wot, bo, bufB, nullptr, 16384, 512, 512, 0);
  ln_resid<<<dim3(2048), blk, 0, stream>>>(x, bufB, g1, be1, bufC);       // y1 -> bufC
  decomp<<<dim3(8, 8, 32), blk, 0, stream>>>(bufC, bufB, abf);            // h: fp32->bufB, bf16->abf

  // FFN (exact GELU) + LN2 + decomp2
  gemm_bf16<<<dim3(2048 / BN, 128), blk, 0, stream>>>(abf, W1t, b1, nullptr, bigbf, 16384, 2048, 512, 1);
  gemm_bf16<<<g512, blk, 0, stream>>>(bigbf, W2t, b2, bufA, nullptr, 16384, 512, 2048, 0);
  ln_resid<<<dim3(2048), blk, 0, stream>>>(bufB, bufA, g2, be2, bufC);    // y2 -> bufC
  decomp<<<dim3(8, 8, 32), blk, 0, stream>>>(bufC, out, nullptr);
}
